// Generator_37907381354498
// MI455X (gfx1250) — matrix operand order repaired and checked by reading, not compiled, not run on hardware
//
#include <hip/hip_runtime.h>
#include <hip/hip_bf16.h>
#include <math.h>

typedef __bf16 bf16_t;
typedef __attribute__((ext_vector_type(16))) __bf16 v16bf;
typedef __attribute__((ext_vector_type(8)))  float  v8f;

#define B_   64
#define P_   196
#define ENC_ 2048
#define ATT_ 512
#define EMB_ 512
#define GRU_ 512
#define VOC_ 10000
#define T_   21
#define NT_  4   // N-subtiles per wave: each wave owns a 16x64 strip of C

__device__ __forceinline__ float fast_sigmoid(float v) {
    // 1/(1+exp(-v)) with v_rcp_f32 instead of the IEEE divide expansion
    return __builtin_amdgcn_rcpf(1.0f + __expf(-v));
}

// ---------------------------------------------------------------------------
// WMMA GEMM:  C[M,N] = act( A[M,K] @ W[K,N] + bias[N] )
//   LDW (W leading dim == N) is a COMPILE-TIME constant so all 16 strided
//   B-element loads per fragment fold into the 24-bit instruction offset of a
//   single address register (max offset 23*10000*4 = 920000 < 2^23).  One
//   pointer per subtile is advanced by 32*LDW per K-step: 4 live address
//   pairs instead of 64 -> no scratch spills, clauses of loads, batched waits.
//   One wave computes a 16x64 strip (4 tiles): one shared A fragment feeds 4
//   independent v_wmma_f32_16x16x32_bf16 chains per K step.
//   ACT: 0 = none, 1 = relu, 2 = sigmoid.
//   M % 16 == 0, K % 32 == 0; partial N strips clamp loads, predicate stores.
// ---------------------------------------------------------------------------
template<int LDW, int ACT>
__global__ void gemm_wmma_bf16(const float* __restrict__ A, int lda,
                               const int* __restrict__ rowOrder, int rowsPerB,
                               const float* __restrict__ W,
                               const float* __restrict__ bias,
                               float* __restrict__ C, long long ldc,
                               int M, int N, int K)
{
    const int lane   = threadIdx.x & 31;
    const int wave   = threadIdx.x >> 5;
    const int tileM  = blockIdx.y;
    const int Ntiles = N >> 4;
    const int tn0    = (blockIdx.x * 8 + wave) * NT_;   // first N-tile of strip
    if (tn0 >= Ntiles) return;

    const int half = lane >> 4;      // which 16-lane half of the wave
    const int l15  = lane & 15;

    // A fragment source row for this lane (M = l15 within the tile)
    int m = tileM * 16 + l15;
    int srcRow = m;
    if (rowOrder) srcRow = rowOrder[m / rowsPerB] * rowsPerB + (m % rowsPerB);
    const float* ap = A + (size_t)srcRow * lda;          // advances by 32/step

    // Column per subtile (clamped in-bounds for loads; stores predicated)
    int  ncol[NT_];
    bool nvalid[NT_];
    const float* wp[NT_];                                // advances by 32*LDW
#pragma unroll
    for (int s = 0; s < NT_; ++s) {
        int tn    = tn0 + s;
        nvalid[s] = (tn < Ntiles);
        ncol[s]   = (nvalid[s] ? tn : tn0) * 16 + l15;
        wp[s]     = W + (size_t)(half * 8) * LDW + ncol[s];
    }

    v8f acc[NT_] = {};
    for (int k0 = 0; k0 < K; k0 += 32) {
        // ---- A fragment: 16x32 bf16.  Lane(l15,half) holds row m; per VGPR j
        //      elements {2j,2j+1} map to K = (j>>2)*16 + half*8 + (j&3)*2 + e.
        v16bf af;
#pragma unroll
        for (int j = 0; j < 8; ++j) {
            int kk = ((j >> 2) << 4) + (half << 3) + ((j & 3) << 1);
            float2 f2 = *(const float2*)(ap + kk);   // 8B aligned: lda, kk even
            af[2 * j]     = (bf16_t)f2.x;
            af[2 * j + 1] = (bf16_t)f2.y;
        }
        // ---- B fragments: 32x16 bf16 each.  Lane(l15,half) holds column n,
        //      K = (i>>3)*16 + half*8 + (i&7); offset folds into the instruction offset.
        v16bf bfrag[NT_];
#pragma unroll
        for (int s = 0; s < NT_; ++s) {
#pragma unroll
            for (int i = 0; i < 16; ++i)
                bfrag[s][i] = (bf16_t)wp[s][(((i >> 3) << 4) + (i & 7)) * LDW];
        }
        // ---- 4 independent WMMAs sharing the A fragment
#pragma unroll
        for (int s = 0; s < NT_; ++s)
            acc[s] = __builtin_amdgcn_wmma_f32_16x16x32_bf16(
                         false, af, false, bfrag[s], (short)0, acc[s], false, false);

        ap += 32;
#pragma unroll
        for (int s = 0; s < NT_; ++s) wp[s] += 32 * LDW;
    }

    // ---- Epilogue: C/D layout — VGPR r holds M = tileM*16 + r + 8*half, N = ncol[s].
#pragma unroll
    for (int s = 0; s < NT_; ++s) {
        if (!nvalid[s]) continue;
        float bv = bias ? bias[ncol[s]] : 0.0f;
#pragma unroll
        for (int r = 0; r < 8; ++r) {
            int   mm = tileM * 16 + r + 8 * half;
            float v  = acc[s][r] + bv;
            if (ACT == 1)      v = fmaxf(v, 0.0f);
            else if (ACT == 2) v = fast_sigmoid(v);
            C[(size_t)mm * (size_t)ldc + ncol[s]] = v;
        }
    }
}

// ---------------------------------------------------------------------------
// Stable descending argsort of cap_lens (rank counting, B=64), gather caps,
// and emit caps_s / output_lens / order output sections (as floats).
// ---------------------------------------------------------------------------
__global__ void order_kernel(const int* __restrict__ cap_lens,
                             const int* __restrict__ caps,
                             int* __restrict__ order_d, int* __restrict__ caps_s,
                             float* __restrict__ out_caps,
                             float* __restrict__ out_lens,
                             float* __restrict__ out_order)
{
    int b   = threadIdx.x;
    int len = cap_lens[b];
    int r   = 0;
    for (int j = 0; j < B_; ++j) {
        int lj = cap_lens[j];
        if (lj > len || (lj == len && j < b)) r++;
    }
    order_d[r] = b;
    __syncthreads();
    int ob = order_d[b];
    for (int j = 0; j < T_; ++j) {
        int c = caps[ob * T_ + j];
        caps_s[b * T_ + j]   = c;
        out_caps[b * T_ + j] = (float)c;
    }
    out_lens[b]  = (float)(cap_lens[ob] - 1);
    out_order[b] = (float)ob;
}

// meanA[b,c] = mean_p img[order[b], p, c]
__global__ void mean_kernel(const float* __restrict__ img,
                            const int* __restrict__ order_d,
                            float* __restrict__ meanA)
{
    int b  = blockIdx.x;
    int ob = order_d[b];
    const float* base = img + (size_t)ob * P_ * ENC_;
    for (int c = threadIdx.x; c < ENC_; c += blockDim.x) {
        float s = 0.0f;
        for (int p = 0; p < P_; ++p) s += base[(size_t)p * ENC_ + c];
        meanA[(size_t)b * ENC_ + c] = s * (1.0f / (float)P_);
    }
}

// xcat[b, 0:512] = emb_table[caps_s[b,t]]
__global__ void emb_kernel(const float* __restrict__ emb_table,
                           const int* __restrict__ caps_s,
                           float* __restrict__ xcat, int t)
{
    int b   = blockIdx.x;
    int tok = caps_s[b * T_ + t];
    const float* e   = emb_table + (size_t)tok * EMB_;
    float*       dst = xcat + (size_t)b * (EMB_ + ENC_);
    for (int j = threadIdx.x; j < EMB_; j += blockDim.x) dst[j] = e[j];
}

// Attention: e = tanh(x1 + y)·Vw + bV ; softmax ; ctx ; xcat[:,512:] = gate*ctx
__global__ void attention_kernel(const float* __restrict__ x1,
                                 const float* __restrict__ y,
                                 const float* __restrict__ Vw,
                                 const float* __restrict__ bVp,
                                 const float* __restrict__ img,
                                 const int* __restrict__ order_d,
                                 const float* __restrict__ gate,
                                 float* __restrict__ alphas_out,
                                 float* __restrict__ xcat, int t)
{
    __shared__ float sh_y[ATT_];
    __shared__ float sh_e[P_];
    __shared__ float red[2];

    int b    = blockIdx.x;
    int tid  = threadIdx.x;
    int lane = tid & 31;
    int wave = tid >> 5;

    for (int a = tid; a < ATT_; a += blockDim.x) sh_y[a] = y[(size_t)b * ATT_ + a];
    __syncthreads();

    float bV = bVp[0];
    for (int p = wave; p < P_; p += 8) {
        const float* xp  = x1 + ((size_t)b * P_ + p) * ATT_;
        float        acc = 0.0f;
        for (int a = lane; a < ATT_; a += 32)
            acc += tanhf(xp[a] + sh_y[a]) * Vw[a];
        for (int off = 16; off > 0; off >>= 1) acc += __shfl_down(acc, off);
        if (lane == 0) sh_e[p] = acc + bV;
    }
    __syncthreads();

    if (tid == 0) {
        float m = sh_e[0];
        for (int p = 1; p < P_; ++p) m = fmaxf(m, sh_e[p]);
        red[0] = m;
    }
    __syncthreads();
    if (tid < P_) sh_e[tid] = __expf(sh_e[tid] - red[0]);
    __syncthreads();
    if (tid == 0) {
        float s = 0.0f;
        for (int p = 0; p < P_; ++p) s += sh_e[p];
        red[1] = __builtin_amdgcn_rcpf(s);
    }
    __syncthreads();
    if (tid < P_) {
        float a   = sh_e[tid] * red[1];
        sh_e[tid] = a;
        alphas_out[((size_t)b * (T_ - 1) + t) * P_ + tid] = a;
    }
    __syncthreads();

    int ob = order_d[b];
    const float* ib  = img  + (size_t)ob * P_ * ENC_;
    const float* gb  = gate + (size_t)b * ENC_;
    float*       dst = xcat + (size_t)b * (EMB_ + ENC_) + EMB_;
    for (int c = tid; c < ENC_; c += blockDim.x) {
        float acc = 0.0f;
        for (int p = 0; p < P_; ++p) acc += sh_e[p] * ib[(size_t)p * ENC_ + c];
        dst[c] = gb[c] * acc;
    }
}

// GRU cell elementwise update (h overwritten with h_new)
__global__ void gru_kernel(const float* __restrict__ gi,
                           const float* __restrict__ gh,
                           float* __restrict__ h)
{
    int idx = blockIdx.x * blockDim.x + threadIdx.x; // 64*512 threads
    int b = idx >> 9, u = idx & 511;
    const float* gib = gi + (size_t)b * 3 * GRU_;
    const float* ghb = gh + (size_t)b * 3 * GRU_;
    float ir = gib[u], iz = gib[GRU_ + u], inn = gib[2 * GRU_ + u];
    float hr = ghb[u], hz = ghb[GRU_ + u], hn  = ghb[2 * GRU_ + u];
    float r = fast_sigmoid(ir + hr);
    float z = fast_sigmoid(iz + hz);
    float n = tanhf(inn + r * hn);
    float hp = h[(size_t)b * GRU_ + u];
    h[(size_t)b * GRU_ + u] = (1.0f - z) * n + z * hp;
}

// ---------------------------------------------------------------------------
template<int LDW, int ACT>
static inline void gemm(const float* A, int lda, const int* ord, int rpb,
                        const float* W, const float* bias,
                        float* C, long long ldc, int M, int N, int K,
                        hipStream_t s)
{
    int ntiles = N / 16;
    dim3 grid((ntiles + 8 * NT_ - 1) / (8 * NT_), M / 16);
    gemm_wmma_bf16<LDW, ACT><<<grid, 256, 0, s>>>(A, lda, ord, rpb, W, bias,
                                                  C, ldc, M, N, K);
}

extern "C" void kernel_launch(void* const* d_in, const int* in_sizes, int n_in,
                              void* d_out, int out_size, void* d_ws, size_t ws_size,
                              hipStream_t stream)
{
    (void)in_sizes; (void)n_in; (void)out_size; (void)ws_size;

    const float* img      = (const float*)d_in[0];
    const int*   caps     = (const int*)  d_in[1];
    const int*   cap_lens = (const int*)  d_in[2];
    const float* W1    = (const float*)d_in[3];
    const float* b1    = (const float*)d_in[4];
    const float* W2    = (const float*)d_in[5];
    const float* b2    = (const float*)d_in[6];
    const float* Vw    = (const float*)d_in[7];
    const float* bV    = (const float*)d_in[8];
    const float* embt  = (const float*)d_in[9];
    const float* Wih   = (const float*)d_in[10];
    const float* bih   = (const float*)d_in[11];
    const float* Whh   = (const float*)d_in[12];
    const float* bhh   = (const float*)d_in[13];
    const float* Winit = (const float*)d_in[14];
    const float* binit = (const float*)d_in[15];
    const float* Wbeta = (const float*)d_in[16];
    const float* bbeta = (const float*)d_in[17];
    const float* Wfc   = (const float*)d_in[18];
    const float* bfc   = (const float*)d_in[19];

    // Output sections (concatenated, return order): preds, caps_s, lens, alphas, order
    float* out        = (float*)d_out;
    float* preds      = out;                                  // 64*20*10000
    float* out_caps   = preds + (size_t)B_ * (T_ - 1) * VOC_; // 64*21
    float* out_lens   = out_caps + B_ * T_;                   // 64
    float* out_alphas = out_lens + B_;                        // 64*20*196
    float* out_order  = out_alphas + (size_t)B_ * (T_ - 1) * P_; // 64

    // Workspace layout (floats)
    float* ws    = (float*)d_ws;
    float* h     = ws;              // 64*512
    float* y     = h     + 32768;   // 64*512
    float* gate  = y     + 32768;   // 64*2048
    float* gh    = gate  + 131072;  // 64*1536
    float* gi    = gh    + 98304;   // 64*1536
    float* xcat  = gi    + 98304;   // 64*2560
    float* meanA = xcat  + 163840;  // 64*2048
    float* x1    = meanA + 131072;  // 12544*512
    int* order_d = (int*)(x1 + (size_t)B_ * P_ * ATT_);
    int* caps_s  = order_d + B_;    // 64*21

    // ---- setup: order, caps_s, lens/order outputs; h0; x1 precompute
    order_kernel<<<1, B_, 0, stream>>>(cap_lens, caps, order_d, caps_s,
                                       out_caps, out_lens, out_order);
    mean_kernel<<<B_, 256, 0, stream>>>(img, order_d, meanA);
    gemm<GRU_, 1>(meanA, ENC_, nullptr, 1, Winit, binit, h, GRU_,
                  B_, GRU_, ENC_, stream);                     // h0 = relu(...)
    gemm<ATT_, 0>(img, ENC_, order_d, P_, W1, b1, x1, ATT_,
                  B_ * P_, ATT_, ENC_, stream);                // x1 precompute

    // ---- recurrent decode loop
    for (int t = 0; t < T_ - 1; ++t) {
        gemm<ATT_, 0>(h, GRU_, nullptr, 1, W2, b2, y, ATT_,
                      B_, ATT_, GRU_, stream);
        gemm<ENC_, 2>(h, GRU_, nullptr, 1, Wbeta, bbeta, gate, ENC_,
                      B_, ENC_, GRU_, stream);                 // sigmoid gate
        gemm<3 * GRU_, 0>(h, GRU_, nullptr, 1, Whh, bhh, gh, 3 * GRU_,
                          B_, 3 * GRU_, GRU_, stream);
        emb_kernel<<<B_, 256, 0, stream>>>(embt, caps_s, xcat, t);
        attention_kernel<<<B_, 256, 0, stream>>>(x1, y, Vw, bV, img, order_d,
                                                 gate, out_alphas, xcat, t);
        gemm<3 * GRU_, 0>(xcat, EMB_ + ENC_, nullptr, 1, Wih, bih, gi, 3 * GRU_,
                          B_, 3 * GRU_, EMB_ + ENC_, stream);
        gru_kernel<<<(B_ * GRU_) / 256, 256, 0, stream>>>(gi, gh, h);
        gemm<VOC_, 0>(h, GRU_, nullptr, 1, Wfc, bfc,
                      preds + (size_t)t * VOC_, (long long)(T_ - 1) * VOC_,
                      B_, VOC_, GRU_, stream);
    }
}
